// MixtureOfExperts_9887014716195
// MI455X (gfx1250) — matrix operand order repaired and checked by reading, not compiled, not run on hardware
//
#include <hip/hip_runtime.h>

#define DIMK   1024
#define NTOK   2048
#define BATCH  8
#define NEXP   8
#define HIDK   2048
#define CAP    512
#define TPE    (BATCH * CAP)   /* 4096 tokens per expert */

typedef __attribute__((ext_vector_type(16))) __bf16 v16bf;
typedef __attribute__((ext_vector_type(8)))  float  v8f;
typedef int i4v __attribute__((__vector_size__(16)));

#define AS1 __attribute__((address_space(1)))
#define AS3 __attribute__((address_space(3)))

#if __has_builtin(__builtin_amdgcn_global_load_async_to_lds_b128) && \
    __has_builtin(__builtin_amdgcn_s_wait_asynccnt)
#define HAVE_ASYNC_LDS 1
#endif

static __device__ __forceinline__ __bf16 f2bf(float f) {
  union { float f; unsigned u; } in; in.f = f;
  unsigned r = (in.u + 0x7FFFu + ((in.u >> 16) & 1u)) >> 16;
  union { unsigned short s; __bf16 b; } out; out.s = (unsigned short)r;
  return out.b;
}

static __device__ __forceinline__ unsigned ballot32(bool p) {
#if __has_builtin(__builtin_amdgcn_ballot_w32)
  return __builtin_amdgcn_ballot_w32(p);
#else
  return (unsigned)__ballot(p);
#endif
}

// ---------------- weight transpose + f32->bf16 convert -------------------
// in: [E][R][C] f32   out: [E][C][R] bf16
__global__ void k_transpose_bf16(const float* __restrict__ in,
                                 __bf16* __restrict__ out, int R, int C) {
  __shared__ float tile[32][33];
  int e = blockIdx.z;
  const float* src = in + (size_t)e * R * C;
  __bf16* dst = out + (size_t)e * R * C;
  int c0 = blockIdx.x * 32, r0 = blockIdx.y * 32;
  int tx = threadIdx.x, ty = threadIdx.y;             // 32 x 8
  #pragma unroll
  for (int i = 0; i < 32; i += 8)
    tile[ty + i][tx] = src[(size_t)(r0 + ty + i) * C + c0 + tx];
  __syncthreads();
  #pragma unroll
  for (int i = 0; i < 32; i += 8)
    dst[(size_t)(c0 + ty + i) * R + r0 + tx] = f2bf(tile[tx][ty + i]);
}

// ---------------- gating: one wave per token -----------------------------
__launch_bounds__(256)
__global__ void k_gate(const float* __restrict__ x, const float* __restrict__ wg,
                       const float* __restrict__ rp,
                       int* __restrict__ idx1, int* __restrict__ idx2,
                       float* __restrict__ g1, float* __restrict__ g2,
                       int* __restrict__ keep2, float* __restrict__ probs) {
  int lane = threadIdx.x & 31;
  int tok = blockIdx.x * 8 + (threadIdx.x >> 5);
  const float* xp = x + (size_t)tok * DIMK;
  float acc[8];
  #pragma unroll
  for (int e = 0; e < 8; ++e) acc[e] = 0.f;
  for (int d = lane; d < DIMK; d += 32) {
    float xv = xp[d];
    float4 wlo = ((const float4*)(wg + d * 8))[0];
    float4 whi = ((const float4*)(wg + d * 8))[1];
    acc[0] += xv * wlo.x; acc[1] += xv * wlo.y;
    acc[2] += xv * wlo.z; acc[3] += xv * wlo.w;
    acc[4] += xv * whi.x; acc[5] += xv * whi.y;
    acc[6] += xv * whi.z; acc[7] += xv * whi.w;
  }
  #pragma unroll
  for (int off = 16; off > 0; off >>= 1)
    #pragma unroll
    for (int e = 0; e < 8; ++e) acc[e] += __shfl_xor(acc[e], off, 32);

  if (lane == 0) {
    float m = acc[0];
    #pragma unroll
    for (int e = 1; e < 8; ++e) m = fmaxf(m, acc[e]);
    float p[8], s = 0.f;
    #pragma unroll
    for (int e = 0; e < 8; ++e) { p[e] = expf(acc[e] - m); s += p[e]; }
    float inv = 1.f / s;
    int i1 = 0; float v1 = -1.f;
    #pragma unroll
    for (int e = 0; e < 8; ++e) {
      p[e] *= inv;
      if (p[e] > v1) { v1 = p[e]; i1 = e; }
    }
    int i2 = -1; float v2 = -1.f;
    #pragma unroll
    for (int e = 0; e < 8; ++e)
      if (e != i1 && p[e] > v2) { v2 = p[e]; i2 = e; }
    float den = v1 + v2 + 1e-9f;
    float a1 = v1 / den, a2 = v2 / den;
    idx1[tok] = i1; idx2[tok] = i2;
    g1[tok] = a1;   g2[tok] = a2;
    keep2[tok] = (rp[tok] < (a2 / 0.2f)) ? 1 : 0;
    #pragma unroll
    for (int e = 0; e < 8; ++e) probs[(size_t)tok * 8 + e] = p[e];
  }
}

// ---------------- capacity scan: one wave per (b,e) ----------------------
__global__ void k_scan(const int* __restrict__ idx1, const int* __restrict__ idx2,
                       const int* __restrict__ keep2,
                       const float* __restrict__ g1, const float* __restrict__ g2,
                       float* __restrict__ w1o, int* __restrict__ s1o,
                       float* __restrict__ w2o, int* __restrict__ s2o,
                       int* __restrict__ tok_slot) {
  int e = blockIdx.x & 7, b = blockIdx.x >> 3;
  int lane = threadIdx.x;
  const int base_tok = b * NTOK;
  int* ts = tok_slot + ((size_t)e * BATCH + b) * CAP;
  unsigned lt = (1u << lane) - 1u;

  int cnt1 = 0;
  for (int n0 = 0; n0 < NTOK; n0 += 32) {
    int tok = base_tok + n0 + lane;
    bool m = (idx1[tok] == e);
    unsigned bal = ballot32(m);
    int pos = cnt1 + __popc(bal & lt);
    if (m) {
      bool kept = pos < CAP;
      w1o[tok] = kept ? g1[tok] : 0.f;
      s1o[tok] = kept ? pos : -1;
      if (kept) ts[pos] = tok;
    }
    cnt1 += __popc(bal);
  }
  int m1c = cnt1 < CAP ? cnt1 : CAP;

  int cnt2 = 0;
  for (int n0 = 0; n0 < NTOK; n0 += 32) {
    int tok = base_tok + n0 + lane;
    bool sel = (idx2[tok] == e);
    bool m = sel && (keep2[tok] != 0);
    unsigned bal = ballot32(m);
    int pos = m1c + cnt2 + __popc(bal & lt);
    if (sel) {
      bool kept = m && (pos < CAP);
      w2o[tok] = kept ? g2[tok] : 0.f;
      s2o[tok] = kept ? pos : -1;
      if (kept) ts[pos] = tok;
    }
    cnt2 += __popc(bal);
  }
  int used = m1c + cnt2; if (used > CAP) used = CAP;
  for (int c = used + lane; c < CAP; c += 32) ts[c] = -1;
}

// ---------------- dispatch gather -> bf16 --------------------------------
__launch_bounds__(256)
__global__ void k_dispatch(const float* __restrict__ x,
                           const int* __restrict__ tok_slot,
                           __bf16* __restrict__ Ein) {
  int slot = blockIdx.x;           // e*TPE + b*CAP + c
  int tok = tok_slot[slot];
  __bf16* dst = Ein + (size_t)slot * DIMK;
  int t = threadIdx.x;
  union { __bf16 b[4]; uint2 u; } pk;
  if (tok >= 0) {
    float4 v = ((const float4*)(x + (size_t)tok * DIMK))[t];
    pk.b[0] = f2bf(v.x); pk.b[1] = f2bf(v.y);
    pk.b[2] = f2bf(v.z); pk.b[3] = f2bf(v.w);
  } else {
    pk.u = make_uint2(0u, 0u);
  }
  *(uint2*)(dst + (size_t)t * 4) = pk.u;
}

// ---------------- bf16 WMMA GEMM: C[M,N] = A[M,K] * Bt[N,K]^T ------------
static __device__ __forceinline__ void storeOut(float* p, float v) { *p = v; }
static __device__ __forceinline__ void storeOut(__bf16* p, float v) { *p = f2bf(v); }

template <bool GELU, typename OutT>
__launch_bounds__(256)
__global__ void k_gemm(const __bf16* __restrict__ A, const __bf16* __restrict__ Bt,
                       OutT* __restrict__ C, int M, int N, int K) {
  int e = blockIdx.z;
  A  += (size_t)e * M * K;
  Bt += (size_t)e * N * K;
  C  += (size_t)e * M * N;

  __shared__ __align__(16) __bf16 sA[2][128][40];   // 40 = 32 + pad, 80B rows (16B mult.)
  __shared__ __align__(16) __bf16 sB[2][128][40];

  int tid = threadIdx.x;
  int lane = tid & 31, wid = tid >> 5;
  int wm = wid >> 1, wn = wid & 1;
  int hf = lane >> 4, l15 = lane & 15;
  int tM = blockIdx.y * 128, tN = blockIdx.x * 128;
  int KT = K >> 5;

  v8f acc[2][4];
  #pragma unroll
  for (int i = 0; i < 2; ++i)
    #pragma unroll
    for (int j = 0; j < 4; ++j)
      #pragma unroll
      for (int t = 0; t < 8; ++t) acc[i][j][t] = 0.f;

#if defined(HAVE_ASYNC_LDS)
  // CDNA5 path: global -> LDS without VGPR staging (ASYNCcnt-tracked)
  auto asyncTile = [&](int buf, int kt) {
    int K0 = kt * 32;
    #pragma unroll
    for (int i = 0; i < 2; ++i) {
      int idx = i * 256 + tid;
      int row = idx >> 2, part = idx & 3;
      __builtin_amdgcn_global_load_async_to_lds_b128(
          (AS1 i4v*)(AS1 void*)(void*)(A + (size_t)(tM + row) * K + K0 + part * 8),
          (AS3 i4v*)(AS3 void*)(void*)(&sA[buf][row][part * 8]), 0, 0);
      __builtin_amdgcn_global_load_async_to_lds_b128(
          (AS1 i4v*)(AS1 void*)(void*)(Bt + (size_t)(tN + row) * K + K0 + part * 8),
          (AS3 i4v*)(AS3 void*)(void*)(&sB[buf][row][part * 8]), 0, 0);
    }
  };
  asyncTile(0, 0);
  __builtin_amdgcn_s_wait_asynccnt(0);
  __syncthreads();
#else
  uint4 ra[2], rb[2];
  auto loadG = [&](int kt) {
    int K0 = kt * 32;
    #pragma unroll
    for (int i = 0; i < 2; ++i) {
      int idx = i * 256 + tid;
      int row = idx >> 2, part = idx & 3;
      ra[i] = *(const uint4*)(A  + (size_t)(tM + row) * K + K0 + part * 8);
      rb[i] = *(const uint4*)(Bt + (size_t)(tN + row) * K + K0 + part * 8);
    }
  };
  auto storeL = [&](int buf) {
    #pragma unroll
    for (int i = 0; i < 2; ++i) {
      int idx = i * 256 + tid;
      int row = idx >> 2, part = idx & 3;
      *(uint4*)&sA[buf][row][part * 8] = ra[i];
      *(uint4*)&sB[buf][row][part * 8] = rb[i];
    }
  };
  loadG(0); storeL(0); __syncthreads();
#endif

  for (int kt = 0; kt < KT; ++kt) {
    int buf = kt & 1;
#if defined(HAVE_ASYNC_LDS)
    if (kt + 1 < KT) asyncTile(buf ^ 1, kt + 1);
#else
    if (kt + 1 < KT) loadG(kt + 1);
#endif

    // A fragments: 16x32 bf16, rows striped per ISA layout (two 16B LDS reads)
    v16bf af[2], bfr[4];
    #pragma unroll
    for (int i = 0; i < 2; ++i) {
      int row = wm * 32 + i * 16 + l15;
      union { v16bf v; uint4 q[2]; } u;
      u.q[0] = *(const uint4*)&sA[buf][row][8 * hf];
      u.q[1] = *(const uint4*)&sA[buf][row][16 + 8 * hf];
      af[i] = u.v;
    }
    // B fragments: 32x16 bf16, K striped per ISA layout like A (two 16B LDS reads)
    #pragma unroll
    for (int j = 0; j < 4; ++j) {
      int col = wn * 64 + j * 16 + l15;
      union { v16bf v; uint4 q[2]; } u;
      u.q[0] = *(const uint4*)&sB[buf][col][8 * hf];
      u.q[1] = *(const uint4*)&sB[buf][col][16 + 8 * hf];
      bfr[j] = u.v;
    }
    #pragma unroll
    for (int i = 0; i < 2; ++i)
      #pragma unroll
      for (int j = 0; j < 4; ++j)
        acc[i][j] = __builtin_amdgcn_wmma_f32_16x16x32_bf16(
            false, af[i], false, bfr[j], (short)0, acc[i][j], false, false);

#if defined(HAVE_ASYNC_LDS)
    if (kt + 1 < KT) __builtin_amdgcn_s_wait_asynccnt(0);
#else
    if (kt + 1 < KT) storeL(buf ^ 1);
#endif
    __syncthreads();
  }

  #pragma unroll
  for (int i = 0; i < 2; ++i)
    #pragma unroll
    for (int j = 0; j < 4; ++j)
      #pragma unroll
      for (int t = 0; t < 8; ++t) {
        int row = tM + wm * 32 + i * 16 + hf * 8 + t;
        int col = tN + wn * 64 + j * 16 + l15;
        float v = acc[i][j][t];
        if (GELU) v = 0.5f * v * (1.0f + erff(v * 0.70710678118654752f));
        storeOut(&C[(size_t)row * N + col], v);
      }
}

// ---------------- combine scatter ----------------------------------------
__launch_bounds__(256)
__global__ void k_combine(const float* __restrict__ Eout,
                          const int* __restrict__ idx1, const int* __restrict__ idx2,
                          const int* __restrict__ s1a, const int* __restrict__ s2a,
                          const float* __restrict__ w1a, const float* __restrict__ w2a,
                          float* __restrict__ out) {
  int bn = blockIdx.x;
  int b = bn >> 11;                 // N = 2048
  int t = threadIdx.x;
  float4 r; r.x = r.y = r.z = r.w = 0.f;
  int s1 = s1a[bn];
  if (s1 >= 0) {
    const float4* p = (const float4*)(Eout +
        ((size_t)idx1[bn] * TPE + (size_t)b * CAP + s1) * DIMK);
    float w = w1a[bn]; float4 v = p[t];
    r.x += w * v.x; r.y += w * v.y; r.z += w * v.z; r.w += w * v.w;
  }
  int s2 = s2a[bn];
  if (s2 >= 0) {
    const float4* p = (const float4*)(Eout +
        ((size_t)idx2[bn] * TPE + (size_t)b * CAP + s2) * DIMK);
    float w = w2a[bn]; float4 v = p[t];
    r.x += w * v.x; r.y += w * v.y; r.z += w * v.z; r.w += w * v.w;
  }
  ((float4*)(out + (size_t)bn * DIMK))[t] = r;
}

// ---------------- deterministic loss -------------------------------------
__launch_bounds__(256)
__global__ void k_lossred(const float* __restrict__ probs,
                          const int* __restrict__ idx1, float* __restrict__ prod) {
  int b = blockIdx.x >> 3, e = blockIdx.x & 7;
  int t = threadIdx.x;
  float s = 0.f; int c = 0;
  for (int n = t; n < NTOK; n += 256) {
    int tok = b * NTOK + n;
    s += probs[(size_t)tok * 8 + e];
    c += (idx1[tok] == e) ? 1 : 0;
  }
  __shared__ float ss[256];
  __shared__ int cc[256];
  ss[t] = s; cc[t] = c; __syncthreads();
  for (int o = 128; o > 0; o >>= 1) {
    if (t < o) { ss[t] += ss[t + o]; cc[t] += cc[t + o]; }
    __syncthreads();
  }
  if (t == 0)
    prod[blockIdx.x] = (ss[0] / (float)NTOK) * ((float)cc[0] / (float)NTOK);
}

__global__ void k_lossfin(const float* __restrict__ prod, float* __restrict__ out) {
  if (threadIdx.x == 0) {
    float s = 0.f;
    for (int i = 0; i < BATCH * NEXP; ++i) s += prod[i];
    out[0] = (s / (float)(BATCH * NEXP)) * (float)(NEXP * NEXP) * 0.01f;
  }
}

// -------------------------------------------------------------------------
extern "C" void kernel_launch(void* const* d_in, const int* in_sizes, int n_in,
                              void* d_out, int out_size, void* d_ws, size_t ws_size,
                              hipStream_t stream) {
  (void)in_sizes; (void)n_in; (void)out_size; (void)ws_size;
  const float* x  = (const float*)d_in[0];          // [B,N,DIM]
  const float* wg = (const float*)d_in[1];          // [DIM,E]
  const float* w1 = (const float*)d_in[2];          // [E,DIM,HID]
  const float* w2 = (const float*)d_in[3];          // [E,HID,DIM]
  const float* rp = (const float*)d_in[4];          // [B,N]
  float* out = (float*)d_out;

  char* base = (char*)d_ws;
  size_t off = 0;
  auto alloc = [&](size_t bytes) {
    void* p = base + off;
    off = (off + bytes + 255) & ~(size_t)255;
    return p;
  };
  __bf16* w1T    = (__bf16*)alloc((size_t)NEXP * HIDK * DIMK * 2);
  __bf16* w2T    = (__bf16*)alloc((size_t)NEXP * DIMK * HIDK * 2);
  __bf16* Ein    = (__bf16*)alloc((size_t)NEXP * TPE * DIMK * 2);
  __bf16* hidden = (__bf16*)alloc((size_t)NEXP * TPE * HIDK * 2);
  float*  Eout   = (float*)alloc((size_t)NEXP * TPE * DIMK * 4);
  int*   idx1    = (int*)alloc((size_t)BATCH * NTOK * 4);
  int*   idx2    = (int*)alloc((size_t)BATCH * NTOK * 4);
  int*   keep2   = (int*)alloc((size_t)BATCH * NTOK * 4);
  float* g1      = (float*)alloc((size_t)BATCH * NTOK * 4);
  float* g2      = (float*)alloc((size_t)BATCH * NTOK * 4);
  float* probs   = (float*)alloc((size_t)BATCH * NTOK * 8 * 4);
  float* w1o     = (float*)alloc((size_t)BATCH * NTOK * 4);
  float* w2o     = (float*)alloc((size_t)BATCH * NTOK * 4);
  int*   s1o     = (int*)alloc((size_t)BATCH * NTOK * 4);
  int*   s2o     = (int*)alloc((size_t)BATCH * NTOK * 4);
  int*   tokslot = (int*)alloc((size_t)NEXP * BATCH * CAP * 4);
  float* prod    = (float*)alloc((size_t)BATCH * NEXP * 4);

  // 1) weight convert + transpose (bf16, K-contiguous B operand)
  k_transpose_bf16<<<dim3(HIDK / 32, DIMK / 32, NEXP), dim3(32, 8), 0, stream>>>(
      w1, w1T, DIMK, HIDK);
  k_transpose_bf16<<<dim3(DIMK / 32, HIDK / 32, NEXP), dim3(32, 8), 0, stream>>>(
      w2, w2T, HIDK, DIMK);

  // 2) gating
  k_gate<<<dim3(BATCH * NTOK / 8), dim3(256), 0, stream>>>(
      x, wg, rp, idx1, idx2, g1, g2, keep2, probs);

  // 3) capacity scan (exclusive cumsum semantics)
  k_scan<<<dim3(BATCH * NEXP), dim3(32), 0, stream>>>(
      idx1, idx2, keep2, g1, g2, w1o, s1o, w2o, s2o, tokslot);

  // 4) dispatch gather -> bf16
  k_dispatch<<<dim3(NEXP * TPE), dim3(256), 0, stream>>>(x, tokslot, Ein);

  // 5) expert FFN: GEMM1 + exact GELU -> bf16 hidden; GEMM2 -> f32 out
  k_gemm<true, __bf16><<<dim3(HIDK / 128, TPE / 128, NEXP), dim3(256), 0, stream>>>(
      Ein, w1T, hidden, TPE, HIDK, DIMK);
  k_gemm<false, float><<<dim3(DIMK / 128, TPE / 128, NEXP), dim3(256), 0, stream>>>(
      hidden, w2T, Eout, TPE, DIMK, HIDK);

  // 6) combine
  k_combine<<<dim3(BATCH * NTOK), dim3(256), 0, stream>>>(
      Eout, idx1, idx2, s1o, s2o, w1o, w2o, out);

  // 7) loss (deterministic reduction)
  k_lossred<<<dim3(BATCH * NEXP), dim3(256), 0, stream>>>(probs, idx1, prod);
  k_lossfin<<<dim3(1), dim3(64), 0, stream>>>(
      prod, out + (size_t)BATCH * NTOK * DIMK);
}
